// HG_32753420599618
// MI455X (gfx1250) — matrix operand order repaired and checked by reading, not compiled, not run on hardware
//
#include <hip/hip_runtime.h>
#include <hip/hip_bf16.h>

typedef __bf16 bf16_t;
typedef __attribute__((ext_vector_type(16))) __bf16 bf16x16;
typedef __attribute__((ext_vector_type(8)))  __bf16 bf16x8;
typedef __attribute__((ext_vector_type(8)))  float  f32x8;

#define HID 128
#define WAVES 8
#define EPB (WAVES * 16)   // 128 edges (or nodes) per block

__device__ __forceinline__ float silu_f(float x) {
    return x / (1.0f + __expf(-x));
}

// A fragment: 16x32 bf16 tile, row-major in LDS with leading dim `ld`.
// ISA layout: lanes 0-15 -> M=lane, elems 0..7 = K kb..kb+7, elems 8..15 = K kb+16..kb+23
//             lanes 16-31 -> M=lane-16, elems 0..7 = K kb+8.., elems 8..15 = K kb+24..
__device__ __forceinline__ bf16x16 frag_a(const bf16_t* base, int lane, int kb, int ld) {
    int m  = lane & 15;
    int hs = (lane >> 4) & 1;
    const bf16_t* p = base + m * ld + kb + hs * 8;
    bf16x8 lo = *(const bf16x8*)(p);
    bf16x8 hi = *(const bf16x8*)(p + 16);
    bf16x16 r;
#pragma unroll
    for (int i = 0; i < 8; ++i) { r[i] = lo[i]; r[i + 8] = hi[i]; }
    return r;
}

// B fragment: 32x16 bf16 tile from column-major weights wcm[n*ld + k].
// ISA layout: lane n = lane&15, lanes 0-15 hold K kb..kb+7, kb+16..kb+23; lanes 16-31 hold K kb+8..kb+15, kb+24..kb+31.
__device__ __forceinline__ bf16x16 frag_b(const bf16_t* wcm, int lane, int nb, int kb, int ld) {
    const bf16_t* p = wcm + (nb + (lane & 15)) * ld + kb + ((lane >> 4) & 1) * 8;
    bf16x8 lo = *(const bf16x8*)(p);
    bf16x8 hi = *(const bf16x8*)(p + 16);
    bf16x16 r;
#pragma unroll
    for (int i = 0; i < 8; ++i) { r[i] = lo[i]; r[i + 8] = hi[i]; }
    return r;
}

__global__ void zero_f32_kernel(float* __restrict__ p, long long n) {
    long long i = (long long)blockIdx.x * blockDim.x + threadIdx.x;
    if (i < n) p[i] = 0.0f;
}

// ---------------------------------------------------------------------------
// Edge kernel: fused edge MLP + coord MLP + segment-sum atomics
// ---------------------------------------------------------------------------
__global__ __launch_bounds__(256) void egcl_edge_kernel(
    const float* __restrict__ h, const int* __restrict__ ei,
    const float* __restrict__ coord,
    const float* __restrict__ We1, const float* __restrict__ be1,
    const float* __restrict__ We2, const float* __restrict__ be2,
    const float* __restrict__ Wc1, const float* __restrict__ bc1,
    const float* __restrict__ Wc2,
    float* __restrict__ aggH, float* __restrict__ aggC, float* __restrict__ cnt,
    int nEdges, int nNodes)
{
    __shared__ bf16_t sWe1[HID * 256];     // column-major [n][k], k = 0..255
    __shared__ bf16_t sWe2[HID * HID];     // column-major [n][k]
    __shared__ bf16_t sWc1[HID * HID];     // column-major [n][k]
    __shared__ float  sWe1r[HID];          // We1[256][n] (radial row, rank-1 update)
    __shared__ float  sB1[HID], sB2[HID], sBc1[HID], sWc2v[HID];
    __shared__ bf16_t sA[WAVES][16 * 256]; // per-wave edge_in tile; later reused for m
    __shared__ bf16_t sM[WAVES][16 * HID]; // per-wave intermediate tile
    __shared__ int    sRow[EPB], sCol[EPB];
    __shared__ float  sCd[EPB][3];
    __shared__ float  sRad[EPB];
    __shared__ float  sPhi[EPB];

    const int tid  = threadIdx.x;
    const int lane = tid & 31;
    const int wv   = tid >> 5;
    const int hs   = lane >> 4;
    const int nn   = lane & 15;

    // ---- stage weights (bf16, column-major) ----
    for (int i = tid; i < HID * 256; i += 256) {
        int n = i >> 8, k = i & 255;
        sWe1[i] = (bf16_t)We1[k * HID + n];
    }
    for (int i = tid; i < HID * HID; i += 256) {
        int n = i >> 7, k = i & 127;
        sWe2[i] = (bf16_t)We2[k * HID + n];
        sWc1[i] = (bf16_t)Wc1[k * HID + n];
    }
    if (tid < HID) {
        sWe1r[tid] = We1[256 * HID + tid];
        sB1[tid]   = be1[tid];
        sB2[tid]   = be2[tid];
        sBc1[tid]  = bc1[tid];
        sWc2v[tid] = Wc2[tid];
    }

    // ---- per-edge metadata (lanes 0..15 of each wave) ----
    const int e0 = blockIdx.x * EPB + wv * 16;
    if (lane < 16) {
        int e  = e0 + lane;
        int ee = e < nEdges ? e : (nEdges - 1);
        int r  = ei[ee];
        int c  = ei[nEdges + ee];
        sRow[wv * 16 + lane] = r;
        sCol[wv * 16 + lane] = c;
        float dx = coord[r * 3 + 0] - coord[c * 3 + 0];
        float dy = coord[r * 3 + 1] - coord[c * 3 + 1];
        float dz = coord[r * 3 + 2] - coord[c * 3 + 2];
        float rad = dx * dx + dy * dy + dz * dz;
        float inv = 1.0f / (sqrtf(rad) + 1e-8f);
        sCd[wv * 16 + lane][0] = dx * inv;
        sCd[wv * 16 + lane][1] = dy * inv;
        sCd[wv * 16 + lane][2] = dz * inv;
        sRad[wv * 16 + lane] = rad;
        sPhi[wv * 16 + lane] = 0.0f;
    }
    __syncthreads();

    // ---- build A tile: 16 edges x 256 cols = [h[row] | h[col]] as bf16 ----
    bf16_t* Aw = sA[wv];
    for (int i = lane; i < 16 * 64; i += 32) {       // 64 float4-groups per edge
        int e = i >> 6;
        int g = i & 63;
        int node = (g < 32) ? sRow[wv * 16 + e] : sCol[wv * 16 + e];
        int coff = (g & 31) * 4;
        float4 v = *(const float4*)(h + (long long)node * HID + coff);
        bf16_t* dst = Aw + e * 256 + ((g < 32) ? 0 : HID) + coff;
        dst[0] = (bf16_t)v.x; dst[1] = (bf16_t)v.y;
        dst[2] = (bf16_t)v.z; dst[3] = (bf16_t)v.w;
    }

    bf16_t* Mw = sM[wv];

    // ---- GEMM1: edge_in[16x257] @ We1 -> SiLU -> Mw (K=256 WMMA + rank-1 radial) ----
    {
        bf16x16 afr[8];
#pragma unroll
        for (int kt = 0; kt < 8; ++kt) afr[kt] = frag_a(Aw, lane, kt * 32, 256);
#pragma unroll
        for (int nt = 0; nt < 8; ++nt) {
            f32x8 acc = {};
#pragma unroll
            for (int kt = 0; kt < 8; ++kt) {
                bf16x16 b = frag_b(sWe1, lane, nt * 16, kt * 32, 256);
                acc = __builtin_amdgcn_wmma_f32_16x16x32_bf16(false, afr[kt], false, b,
                                                              (short)0, acc, false, false);
            }
            float wr   = sWe1r[nt * 16 + nn];
            float bias = sB1[nt * 16 + nn];
#pragma unroll
            for (int r = 0; r < 8; ++r) {
                int m = r + 8 * hs;
                float v = acc[r] + sRad[wv * 16 + m] * wr + bias;
                Mw[m * HID + nt * 16 + nn] = (bf16_t)silu_f(v);
            }
        }
    }

    // ---- GEMM2: Mw @ We2 -> SiLU -> message m (stored in Aw, ld=128) ----
    {
        bf16x16 afr[4];
#pragma unroll
        for (int kt = 0; kt < 4; ++kt) afr[kt] = frag_a(Mw, lane, kt * 32, HID);
#pragma unroll
        for (int nt = 0; nt < 8; ++nt) {
            f32x8 acc = {};
#pragma unroll
            for (int kt = 0; kt < 4; ++kt) {
                bf16x16 b = frag_b(sWe2, lane, nt * 16, kt * 32, HID);
                acc = __builtin_amdgcn_wmma_f32_16x16x32_bf16(false, afr[kt], false, b,
                                                              (short)0, acc, false, false);
            }
            float bias = sB2[nt * 16 + nn];
#pragma unroll
            for (int r = 0; r < 8; ++r) {
                int m = r + 8 * hs;
                Aw[m * HID + nt * 16 + nn] = (bf16_t)silu_f(acc[r] + bias);
            }
        }
    }

    // ---- GEMM3: m @ Wc1 -> SiLU -> dot with Wc2 => phi_x per edge ----
    {
        float phi_part[8];
#pragma unroll
        for (int r = 0; r < 8; ++r) phi_part[r] = 0.0f;
        bf16x16 afr[4];
#pragma unroll
        for (int kt = 0; kt < 4; ++kt) afr[kt] = frag_a(Aw, lane, kt * 32, HID);
#pragma unroll
        for (int nt = 0; nt < 8; ++nt) {
            f32x8 acc = {};
#pragma unroll
            for (int kt = 0; kt < 4; ++kt) {
                bf16x16 b = frag_b(sWc1, lane, nt * 16, kt * 32, HID);
                acc = __builtin_amdgcn_wmma_f32_16x16x32_bf16(false, afr[kt], false, b,
                                                              (short)0, acc, false, false);
            }
            float bias = sBc1[nt * 16 + nn];
            float w2   = sWc2v[nt * 16 + nn];
#pragma unroll
            for (int r = 0; r < 8; ++r)
                phi_part[r] += silu_f(acc[r] + bias) * w2;
        }
#pragma unroll
        for (int r = 0; r < 8; ++r)
            atomicAdd(&sPhi[wv * 16 + r + 8 * hs], phi_part[r]);
    }

    // ---- coord aggregation atomics (mean numerator + degree count) ----
    if (lane < 16) {
        int e = e0 + lane;
        if (e < nEdges) {
            int r    = sRow[wv * 16 + lane];
            float ph = sPhi[wv * 16 + lane];
            atomicAdd(&aggC[r * 3 + 0], sCd[wv * 16 + lane][0] * ph);
            atomicAdd(&aggC[r * 3 + 1], sCd[wv * 16 + lane][1] * ph);
            atomicAdd(&aggC[r * 3 + 2], sCd[wv * 16 + lane][2] * ph);
            atomicAdd(&cnt[r], 1.0f);
        }
    }

    // ---- message aggregation atomics: agg_h[row] += m ----
    for (int i = lane; i < 16 * HID; i += 32) {
        int e  = i >> 7;
        int c  = i & 127;
        int ge = e0 + e;
        if (ge < nEdges) {
            atomicAdd(&aggH[(long long)sRow[wv * 16 + e] * HID + c], (float)Aw[e * HID + c]);
        }
    }
}

// ---------------------------------------------------------------------------
// Node kernel: coord update + node MLP with residual
// ---------------------------------------------------------------------------
__global__ __launch_bounds__(256) void egcl_node_kernel(
    const float* __restrict__ h, const float* __restrict__ coord,
    const float* __restrict__ Wn1, const float* __restrict__ bn1,
    const float* __restrict__ Wn2, const float* __restrict__ bn2,
    const float* __restrict__ aggH, const float* __restrict__ aggC,
    const float* __restrict__ cnt,
    float* __restrict__ hOut, float* __restrict__ cOut, int nNodes)
{
    __shared__ bf16_t sWn1[HID * 256];     // column-major [n][k], k = 0..255
    __shared__ bf16_t sWn2[HID * HID];     // column-major [n][k]
    __shared__ float  sB1[HID], sB2[HID];
    __shared__ bf16_t sA[WAVES][16 * 256];
    __shared__ bf16_t sT[WAVES][16 * HID];

    const int tid  = threadIdx.x;
    const int lane = tid & 31;
    const int wv   = tid >> 5;
    const int hs   = lane >> 4;
    const int nn   = lane & 15;

    for (int i = tid; i < HID * 256; i += 256) {
        int n = i >> 8, k = i & 255;
        sWn1[i] = (bf16_t)Wn1[k * HID + n];
    }
    for (int i = tid; i < HID * HID; i += 256) {
        int n = i >> 7, k = i & 127;
        sWn2[i] = (bf16_t)Wn2[k * HID + n];
    }
    if (tid < HID) { sB1[tid] = bn1[tid]; sB2[tid] = bn2[tid]; }

    const int nb = blockIdx.x * EPB + wv * 16;

    // ---- coord_out = coord + agg_c / max(cnt, 1) ----
    if (lane < 16) {
        int nd = nb + lane;
        if (nd < nNodes) {
            float c = fmaxf(cnt[nd], 1.0f);
            float invc = 1.0f / c;
#pragma unroll
            for (int j = 0; j < 3; ++j)
                cOut[nd * 3 + j] = coord[nd * 3 + j] + aggC[nd * 3 + j] * invc;
        }
    }

    // ---- build A tile: [h | agg_h] (16 nodes x 256) ----
    bf16_t* Aw = sA[wv];
    for (int i = lane; i < 16 * 64; i += 32) {
        int e = i >> 6;
        int g = i & 63;
        int nd = nb + e;
        if (nd >= nNodes) nd = nNodes - 1;
        int coff = (g & 31) * 4;
        const float* src = (g < 32) ? (h + (long long)nd * HID + coff)
                                    : (aggH + (long long)nd * HID + coff);
        float4 v = *(const float4*)src;
        bf16_t* dst = Aw + e * 256 + ((g < 32) ? 0 : HID) + coff;
        dst[0] = (bf16_t)v.x; dst[1] = (bf16_t)v.y;
        dst[2] = (bf16_t)v.z; dst[3] = (bf16_t)v.w;
    }
    __syncthreads();

    bf16_t* Tw = sT[wv];

    // ---- GEMM1: node_in @ Wn1 -> SiLU ----
    {
        bf16x16 afr[8];
#pragma unroll
        for (int kt = 0; kt < 8; ++kt) afr[kt] = frag_a(Aw, lane, kt * 32, 256);
#pragma unroll
        for (int nt = 0; nt < 8; ++nt) {
            f32x8 acc = {};
#pragma unroll
            for (int kt = 0; kt < 8; ++kt) {
                bf16x16 b = frag_b(sWn1, lane, nt * 16, kt * 32, 256);
                acc = __builtin_amdgcn_wmma_f32_16x16x32_bf16(false, afr[kt], false, b,
                                                              (short)0, acc, false, false);
            }
            float bias = sB1[nt * 16 + nn];
#pragma unroll
            for (int r = 0; r < 8; ++r) {
                int m = r + 8 * hs;
                Tw[m * HID + nt * 16 + nn] = (bf16_t)silu_f(acc[r] + bias);
            }
        }
    }

    // ---- GEMM2: @ Wn2 + bn2 + h (residual) -> h_out ----
    {
        bf16x16 afr[4];
#pragma unroll
        for (int kt = 0; kt < 4; ++kt) afr[kt] = frag_a(Tw, lane, kt * 32, HID);
#pragma unroll
        for (int nt = 0; nt < 8; ++nt) {
            f32x8 acc = {};
#pragma unroll
            for (int kt = 0; kt < 4; ++kt) {
                bf16x16 b = frag_b(sWn2, lane, nt * 16, kt * 32, HID);
                acc = __builtin_amdgcn_wmma_f32_16x16x32_bf16(false, afr[kt], false, b,
                                                              (short)0, acc, false, false);
            }
            float bias = sB2[nt * 16 + nn];
#pragma unroll
            for (int r = 0; r < 8; ++r) {
                int m  = r + 8 * hs;
                int nd = nb + m;
                int n  = nt * 16 + nn;
                if (nd < nNodes)
                    hOut[(long long)nd * HID + n] = acc[r] + bias + h[(long long)nd * HID + n];
            }
        }
    }
}

// ---------------------------------------------------------------------------
extern "C" void kernel_launch(void* const* d_in, const int* in_sizes, int n_in,
                              void* d_out, int out_size, void* d_ws, size_t ws_size,
                              hipStream_t stream) {
    const float* h     = (const float*)d_in[0];
    const int*   ei    = (const int*)d_in[1];
    const float* coord = (const float*)d_in[2];
    const float* We1   = (const float*)d_in[3];
    const float* be1   = (const float*)d_in[4];
    const float* We2   = (const float*)d_in[5];
    const float* be2   = (const float*)d_in[6];
    const float* Wn1   = (const float*)d_in[7];
    const float* bn1   = (const float*)d_in[8];
    const float* Wn2   = (const float*)d_in[9];
    const float* bn2   = (const float*)d_in[10];
    const float* Wc1   = (const float*)d_in[11];
    const float* bc1   = (const float*)d_in[12];
    const float* Wc2   = (const float*)d_in[13];

    const int N = in_sizes[0] / HID;
    const int E = in_sizes[1] / 2;

    float* ws   = (float*)d_ws;
    float* aggH = ws;                                 // N*128
    float* aggC = ws + (long long)N * HID;            // N*3
    float* cnt  = aggC + (long long)N * 3;            // N

    float* hOut = (float*)d_out;
    float* cOut = hOut + (long long)N * HID;

    long long zn = (long long)N * (HID + 4);
    zero_f32_kernel<<<(int)((zn + 255) / 256), 256, 0, stream>>>(ws, zn);

    int edgeBlocks = (E + EPB - 1) / EPB;
    egcl_edge_kernel<<<edgeBlocks, 256, 0, stream>>>(
        h, ei, coord, We1, be1, We2, be2, Wc1, bc1, Wc2,
        aggH, aggC, cnt, E, N);

    int nodeBlocks = (N + EPB - 1) / EPB;
    egcl_node_kernel<<<nodeBlocks, 256, 0, stream>>>(
        h, coord, Wn1, bn1, Wn2, bn2, aggH, aggC, cnt, hOut, cOut, N);
}
